// LocalMaxSA_21852793602393
// MI455X (gfx1250) — compile-verified, matrix operand order checked by reading, not run on hardware
//
#include <hip/hip_runtime.h>
#include <math.h>

typedef __bf16 bf16;
typedef __attribute__((ext_vector_type(16))) __bf16 v16bf;
typedef __attribute__((ext_vector_type(8)))  __bf16 bf16x8;
typedef __attribute__((ext_vector_type(8)))  float  v8f;

#define N_FEAT 256
#define SEQ_T  2048
#define N_B    8
#define BT     (N_B * SEQ_T)

__device__ __forceinline__ v8f wmma_bf16(v16bf a, v16bf b, v8f c) {
    // D = A(16x32 bf16) * B(32x16 bf16) + C(16x16 f32)
    return __builtin_amdgcn_wmma_f32_16x16x32_bf16(
        /*neg_a=*/false, a, /*neg_b=*/false, b,
        /*c_mod=*/(short)0, c, /*reuse_a=*/false, /*reuse_b=*/false);
}

// Load a 16x32 (rows x K) bf16 fragment from a row-major matrix.
// Per ISA layout: lane holds row (row0 + lane%16); K chunks at
// k0 + 8*(lane/16) and k0 + 16 + 8*(lane/16), 8 contiguous bf16 each (16B).
__device__ __forceinline__ v16bf load_frag_bf16(const bf16* __restrict__ base,
                                                int ld, int row0, int k0) {
    int lane = threadIdx.x & 31;
    int lo = lane & 15, hi = lane >> 4;
    const bf16* p = base + (size_t)(row0 + lo) * ld + k0 + hi * 8;
    bf16x8 c0 = *(const bf16x8*)p;
    bf16x8 c1 = *(const bf16x8*)(p + 16);
    v16bf f;
#pragma unroll
    for (int i = 0; i < 8; ++i) { f[i] = c0[i]; f[i + 8] = c1[i]; }
    return f;
}

// Same fragment, but source is fp32 row-major (converted to bf16 on the fly).
__device__ __forceinline__ v16bf load_frag_f32(const float* __restrict__ base,
                                               int ld, int row0, int k0) {
    int lane = threadIdx.x & 31;
    int lo = lane & 15, hi = lane >> 4;
    const float* p = base + (size_t)(row0 + lo) * ld + k0 + hi * 8;
    v16bf f;
#pragma unroll
    for (int i = 0; i < 8; ++i) { f[i] = (bf16)p[i]; f[i + 8] = (bf16)p[16 + i]; }
    return f;
}

// ---------------------------------------------------------------------------
// Kernel 1: q = feat*Wq^T + bq ; k = feat*Wk^T + bk  (bf16 outputs, row-major)
// Per wave: 16 rows x 32 output cols, for both q and k.
// ---------------------------------------------------------------------------
__global__ void proj_qk_kernel(const float* __restrict__ feat,
                               const float* __restrict__ Wq,
                               const float* __restrict__ bq,
                               const float* __restrict__ Wk,
                               const float* __restrict__ bk,
                               bf16* __restrict__ qbf,
                               bf16* __restrict__ kbf) {
    int wid = blockIdx.x * (blockDim.x >> 5) + (threadIdx.x >> 5);
    int rt = wid >> 3;   // 0..1023 row tile (16 rows of B*T)
    int gp = wid & 7;    // 0..7 column pair (32 cols)
    int r0 = rt * 16, g0 = gp * 32;
    int lane = threadIdx.x & 31, lo = lane & 15, hi = lane >> 4;

    v8f aq0 = {}, aq1 = {}, ak0 = {}, ak1 = {};
#pragma unroll
    for (int kk = 0; kk < 8; ++kk) {
        int k0 = kk * 32;
        v16bf af  = load_frag_f32(feat, N_FEAT, r0, k0);       // A = feat rows
        v16bf bq0 = load_frag_f32(Wq, N_FEAT, g0,      k0);    // B = Wq^T cols = Wq rows
        v16bf bq1 = load_frag_f32(Wq, N_FEAT, g0 + 16, k0);
        v16bf bk0 = load_frag_f32(Wk, N_FEAT, g0,      k0);
        v16bf bk1 = load_frag_f32(Wk, N_FEAT, g0 + 16, k0);
        aq0 = wmma_bf16(af, bq0, aq0);
        aq1 = wmma_bf16(af, bq1, aq1);
        ak0 = wmma_bf16(af, bk0, ak0);
        ak1 = wmma_bf16(af, bk1, ak1);
    }
    float biasq0 = bq[g0 + lo], biasq1 = bq[g0 + 16 + lo];
    float biask0 = bk[g0 + lo], biask1 = bk[g0 + 16 + lo];
#pragma unroll
    for (int v = 0; v < 8; ++v) {
        int row = r0 + v + 8 * hi;   // D layout: vgpr v -> row v + 8*(lane/16)
        qbf[(size_t)row * N_FEAT + g0 + lo]      = (bf16)(aq0[v] + biasq0);
        qbf[(size_t)row * N_FEAT + g0 + 16 + lo] = (bf16)(aq1[v] + biasq1);
        kbf[(size_t)row * N_FEAT + g0 + lo]      = (bf16)(ak0[v] + biask0);
        kbf[(size_t)row * N_FEAT + g0 + 16 + lo] = (bf16)(ak1[v] + biask1);
    }
}

// ---------------------------------------------------------------------------
// Kernel 2: featT[b][f][s] = (bf16) feat[b][s][f]  (LDS-tiled transpose)
// Also writes hlens_new = T (constant).
// ---------------------------------------------------------------------------
__global__ void transpose_kernel(const float* __restrict__ feat,
                                 bf16* __restrict__ featT,
                                 float* __restrict__ hlens_out) {
    __shared__ float tile[32][33];
    int b = blockIdx.z;
    int s0 = blockIdx.x * 32, f0 = blockIdx.y * 32;
    int tx = threadIdx.x, ty = threadIdx.y;  // (32, 8)
    const float* src = feat + (size_t)b * SEQ_T * N_FEAT;
#pragma unroll
    for (int i = 0; i < 4; ++i) {
        int s = ty + i * 8;
        tile[s][tx] = src[(size_t)(s0 + s) * N_FEAT + f0 + tx];
    }
    __syncthreads();
    bf16* dst = featT + (size_t)b * N_FEAT * SEQ_T;
#pragma unroll
    for (int i = 0; i < 4; ++i) {
        int f = ty + i * 8;
        dst[(size_t)(f0 + f) * SEQ_T + s0 + tx] = (bf16)tile[tx][f];
    }
    if (blockIdx.x == 0 && blockIdx.y == 0 && blockIdx.z == 0 &&
        threadIdx.x == 0 && threadIdx.y == 0) {
        for (int i = 0; i < N_B; ++i) hlens_out[i] = (float)SEQ_T;
    }
}

// ---------------------------------------------------------------------------
// Kernel 3: fused attention. Per wave: 16 query rows x 128 output features
// (two waves cover one 16-row tile -> 2048 waves, ~180 VGPRs peak, no spill).
//   q-fragments (loop invariant) are hoisted into 64 VGPRs.
//   Pass 1: S^T = k * q^T tiles -> row max & raw sum (score).
//   Pass 2: recompute S^T, P = exp(S - m), O += P * feat; normalize by l.
// Computing S TRANSPOSED makes softmax reductions and the P->A-fragment
// conversion lane-local (one shfl_xor(16) at the end of each pass).
// ---------------------------------------------------------------------------
__global__ void attn_kernel(const bf16* __restrict__ qbf,
                            const bf16* __restrict__ kbf,
                            const bf16* __restrict__ featT,
                            float* __restrict__ out_feat,
                            float* __restrict__ out_score) {
    int wid = blockIdx.x * (blockDim.x >> 5) + (threadIdx.x >> 5);
    int tile_id = wid >> 1;          // 0..1023: (batch, t-tile)
    int fhalf   = wid & 1;           // which 128-feature half this wave owns
    int b  = tile_id >> 7;           // 128 row-tiles per batch
    int t0 = (tile_id & 127) * 16;
    int f_base = fhalf * 128;
    int lane = threadIdx.x & 31, lo = lane & 15, hi = lane >> 4;

    const bf16* qb  = qbf   + (size_t)b * SEQ_T * N_FEAT;
    const bf16* kb  = kbf   + (size_t)b * SEQ_T * N_FEAT;
    const bf16* ftb = featT + (size_t)b * N_FEAT * SEQ_T;

    // Hoist the 16x256 q-block fragments (B operands): 8 x v16bf = 64 VGPRs.
    v16bf qf[8];
#pragma unroll
    for (int kk = 0; kk < 8; ++kk) qf[kk] = load_frag_bf16(qb, N_FEAT, t0, kk * 32);

    // ---- pass 1: running max + raw sum over s (lane covers t=lo, s=v+8*hi+s0)
    float m = -INFINITY, r = 0.0f;
    for (int s0 = 0; s0 < SEQ_T; s0 += 16) {
        v8f sacc = {};
#pragma unroll
        for (int kk = 0; kk < 8; ++kk) {
            v16bf ak = load_frag_bf16(kb, N_FEAT, s0, kk * 32);  // A = k rows
            sacc = wmma_bf16(ak, qf[kk], sacc);
        }
#pragma unroll
        for (int v = 0; v < 8; ++v) { m = fmaxf(m, sacc[v]); r += sacc[v]; }
    }
    m = fmaxf(m, __shfl_xor(m, 16));
    r += __shfl_xor(r, 16);
    if (fhalf == 0 && hi == 0)
        out_score[(size_t)b * SEQ_T + t0 + lo] = r * (1.0f / SEQ_T);

    // ---- pass 2: P = exp(S - m); O = P * feat (this wave's 8 f-tiles)
    v8f acc[8];
#pragma unroll
    for (int ff = 0; ff < 8; ++ff) acc[ff] = (v8f){};
    float l = 0.0f;

    for (int s0 = 0; s0 < SEQ_T; s0 += 32) {
        v16bf pa;  // A-fragment of P (16t x 32s), built lane-locally from S^T
#pragma unroll
        for (int j = 0; j < 2; ++j) {
            v8f sacc = {};
#pragma unroll
            for (int kk = 0; kk < 8; ++kk) {
                v16bf ak = load_frag_bf16(kb, N_FEAT, s0 + 16 * j, kk * 32);
                sacc = wmma_bf16(ak, qf[kk], sacc);
            }
#pragma unroll
            for (int v = 0; v < 8; ++v) {
                float p = __expf(sacc[v] - m);
                l += p;
                // S^T D-layout (lane: t=lo, s=v+8*hi) == A-layout slot j*8+v
                pa[j * 8 + v] = (bf16)p;
            }
        }
#pragma unroll
        for (int ff = 0; ff < 8; ++ff) {
            // B = feat[s][f] tile, loaded contiguously from featT rows
            v16bf bfr = load_frag_bf16(ftb, SEQ_T, f_base + ff * 16, s0);
            acc[ff] = wmma_bf16(pa, bfr, acc[ff]);
        }
    }
    l += __shfl_xor(l, 16);
    float linv = 1.0f / l;
    float lrow[8];
#pragma unroll
    for (int v = 0; v < 8; ++v) lrow[v] = __shfl(linv, v + 8 * hi);

    float* ob = out_feat + (size_t)b * SEQ_T * N_FEAT;
#pragma unroll
    for (int ff = 0; ff < 8; ++ff) {
#pragma unroll
        for (int v = 0; v < 8; ++v) {
            int row = t0 + v + 8 * hi;
            ob[(size_t)row * N_FEAT + f_base + ff * 16 + lo] = acc[ff][v] * lrow[v];
        }
    }
}

// ---------------------------------------------------------------------------
extern "C" void kernel_launch(void* const* d_in, const int* in_sizes, int n_in,
                              void* d_out, int out_size, void* d_ws, size_t ws_size,
                              hipStream_t stream) {
    const float* feat = (const float*)d_in[0];
    // d_in[1] = hlens (unused: reference overwrites with T)
    const float* Wq = (const float*)d_in[2];
    const float* bq = (const float*)d_in[3];
    const float* Wk = (const float*)d_in[4];
    const float* bk = (const float*)d_in[5];

    float* out_feat  = (float*)d_out;                         // [8,2048,256]
    float* out_hlens = out_feat + (size_t)BT * N_FEAT;        // [8]
    float* out_score = out_hlens + N_B;                       // [8,2048]

    bf16* qbf   = (bf16*)d_ws;                                // 8 MB
    bf16* kbf   = qbf + (size_t)BT * N_FEAT;                  // 8 MB
    bf16* featT = kbf + (size_t)BT * N_FEAT;                  // 8 MB

    // Kernel 1: projections. 1024 row-tiles x 8 col-pairs = 8192 waves.
    proj_qk_kernel<<<2048, 128, 0, stream>>>(feat, Wq, bq, Wk, bk, qbf, kbf);

    // Kernel 2: transpose feat -> featT (bf16), write hlens_new.
    transpose_kernel<<<dim3(SEQ_T / 32, N_FEAT / 32, N_B), dim3(32, 8), 0, stream>>>(
        feat, featT, out_hlens);

    // Kernel 3: fused attention. 1024 tiles x 2 f-halves = 2048 waves.
    attn_kernel<<<512, 128, 0, stream>>>(qbf, kbf, featT, out_feat, out_score);
}
